// SoftMoELayerWrapper_26121991094683
// MI455X (gfx1250) — compile-verified, matrix operand order checked by reading, not run on hardware
//
#include <hip/hip_runtime.h>

// ---------------------------------------------------------------------------
// Soft-MoE forward for MI455X (gfx1250, wave32, WMMA bf16 16x16x32, f32 accum)
// B=8, M=4096, D=1024, N=64, P=2  (NP=128)
// Pipeline:
//   prep:   xn_bf16 (L2-normalized rows of x), xT_bf16 (raw x transposed),
//           phin_bf16 (scale * normalized phi, stored [slot][D]),
//           Wt_bf16 (W[n] transposed, [n][e][d])
//   GEMM1:  logits = xn @ phin^T / TEMP              (WMMA)
//   softmax: c over slots (128), d over tokens (4096)
//   GEMM2:  xs = d^T @ x                              (WMMA)
//   GEMM3:  ys = xs @ W[n] + b   -> stored transposed (WMMA)
//   GEMM4:  y  = c @ ys + x (residual)                (WMMA)
// GEMM loops batch all operand loads of a k-step ahead of a multi-WMMA batch
// (one s_clause, one wait, several v_wmma). No cross-iteration register
// rotation: that spilled to scratch in the previous revision.
// ---------------------------------------------------------------------------

#define B_   8
#define M_   4096
#define D_   1024
#define N_   64
#define NP_  128
#define TEMP_INV 10.0f
#define EPS_ 1e-12f

typedef __attribute__((ext_vector_type(16))) __bf16 v16bf;
typedef __attribute__((ext_vector_type(8)))  __bf16 v8bf;
typedef __attribute__((ext_vector_type(8)))  float  v8f;

#define WMMA_BF16(a, b, c) \
  __builtin_amdgcn_wmma_f32_16x16x32_bf16(false, (a), false, (b), (short)0, (c), false, false)

// float -> bf16, round-to-nearest-even (bit math; avoids relying on cast codegen)
__device__ __forceinline__ __bf16 f2bf(float f) {
  union { float f; unsigned u; } in; in.f = f;
  unsigned u = in.u;
  u += 0x7fffu + ((u >> 16) & 1u);
  unsigned short h = (unsigned short)(u >> 16);
  union { unsigned short s; __bf16 b; } out; out.s = h;
  return out.b;
}

// Load one wave-lane's slice of a 16x32 (MxK) bf16 operand tile.
// ISA layout: lane holds K = [k0+8*half .. +7] in elems 0..7 and
//             K = [k0+16+8*half .. +7] in elems 8..15, for its row.
// Works for both A (row = M index) and B (row = N index, operand stored
// transposed so K is contiguous in memory).
__device__ __forceinline__ v16bf load_k32(const __bf16* __restrict__ base,
                                          int ld, int row, int k0, int kh) {
  const __bf16* p = base + (size_t)row * ld + k0 + 8 * kh;
  v8bf lo = *(const v8bf*)(p);
  v8bf hi = *(const v8bf*)(p + 16);
  v16bf r;
#pragma unroll
  for (int i = 0; i < 8; ++i) { r[i] = lo[i]; r[i + 8] = hi[i]; }
  return r;
}

// ---------------- prep kernels -------------------------------------------

// One block per token row: L2-normalize over D, emit bf16 row.
__global__ void smoe_norm_x(const float* __restrict__ x, __bf16* __restrict__ xn) {
  int row = blockIdx.x, t = threadIdx.x;                 // 256 threads
  const float4 v = ((const float4*)(x + (size_t)row * D_))[t];
  __shared__ float red[256];
  red[t] = v.x * v.x + v.y * v.y + v.z * v.z + v.w * v.w;
  __syncthreads();
  for (int s = 128; s > 0; s >>= 1) { if (t < s) red[t] += red[t + s]; __syncthreads(); }
  float inv = 1.0f / fmaxf(sqrtf(red[0]), EPS_);
  __bf16* o = xn + (size_t)row * D_ + 4 * t;
  o[0] = f2bf(v.x * inv); o[1] = f2bf(v.y * inv);
  o[2] = f2bf(v.z * inv); o[3] = f2bf(v.w * inv);
}

// One block per slot s: normalize phi[:, s] over D, apply scale, store [s][D].
__global__ void smoe_prep_phi(const float* __restrict__ phi, const float* __restrict__ scale,
                              __bf16* __restrict__ phin) {
  int s = blockIdx.x, t = threadIdx.x;                   // 128 blocks x 256 thr
  float v[4], ss = 0.f;
#pragma unroll
  for (int i = 0; i < 4; ++i) { v[i] = phi[(size_t)(t + 256 * i) * NP_ + s]; ss += v[i] * v[i]; }
  __shared__ float red[256];
  red[t] = ss; __syncthreads();
  for (int q = 128; q > 0; q >>= 1) { if (t < q) red[t] += red[t + q]; __syncthreads(); }
  float sc = scale[0] / fmaxf(sqrtf(red[0]), EPS_);
#pragma unroll
  for (int i = 0; i < 4; ++i) phin[(size_t)s * D_ + t + 256 * i] = f2bf(v[i] * sc);
}

// Tiled transpose: per-batch x[M][D] f32 -> xT[D][M] bf16 (raw x, for GEMM2 B).
__global__ void smoe_tr_x(const float* __restrict__ x, __bf16* __restrict__ xT) {
  __shared__ float tile[32][33];
  int b = blockIdx.z, m0 = blockIdx.x * 32, d0 = blockIdx.y * 32;
  int tx = threadIdx.x, ty = threadIdx.y;                // 32x8
  const float* xb = x + (size_t)b * M_ * D_;
#pragma unroll
  for (int i = 0; i < 4; ++i)
    tile[ty + 8 * i][tx] = xb[(size_t)(m0 + ty + 8 * i) * D_ + d0 + tx];
  __syncthreads();
  __bf16* ob = xT + (size_t)b * D_ * M_;
#pragma unroll
  for (int i = 0; i < 4; ++i)
    ob[(size_t)(d0 + ty + 8 * i) * M_ + m0 + tx] = f2bf(tile[tx][ty + 8 * i]);
}

// Tiled transpose: per-expert W[n][d][e] f32 -> Wt[n][e][d] bf16 (GEMM3 B).
__global__ void smoe_tr_w(const float* __restrict__ W, __bf16* __restrict__ Wt) {
  __shared__ float tile[32][33];
  int n = blockIdx.z, d0 = blockIdx.x * 32, e0 = blockIdx.y * 32;
  int tx = threadIdx.x, ty = threadIdx.y;                // 32x8
  const float* Wn = W + (size_t)n * D_ * D_;
#pragma unroll
  for (int i = 0; i < 4; ++i)
    tile[ty + 8 * i][tx] = Wn[(size_t)(d0 + ty + 8 * i) * D_ + e0 + tx];
  __syncthreads();
  __bf16* on = Wt + (size_t)n * D_ * D_;
#pragma unroll
  for (int i = 0; i < 4; ++i)
    on[(size_t)(e0 + ty + 8 * i) * D_ + d0 + tx] = f2bf(tile[tx][ty + 8 * i]);
}

// ---------------- GEMM1: logits = xn @ phin^T / TEMP ----------------------
// Block = 4 waves covering 2 row-tiles; each wave: 16 rows x 64 cols, K=1024.
// Per k-step: 10-load clause (A + 4 B tiles), then 4 WMMAs.
__global__ void smoe_logits(const __bf16* __restrict__ xn, const __bf16* __restrict__ phin,
                            float* __restrict__ logits) {
  int wave = threadIdx.x >> 5, lane = threadIdx.x & 31;
  int ln = lane & 15, kh = lane >> 4;
  int rowtile = blockIdx.x * 2 + (wave >> 1);
  int c0 = (wave & 1) * 4;                               // column tiles c0..c0+3
  const __bf16* A = xn + (size_t)rowtile * 16 * D_;
  v8f zero = {};
  v8f acc[4];
#pragma unroll
  for (int c = 0; c < 4; ++c) acc[c] = zero;
  for (int k0 = 0; k0 < D_; k0 += 32) {
    v16bf a = load_k32(A, D_, ln, k0, kh);
    v16bf b[4];
#pragma unroll
    for (int c = 0; c < 4; ++c) b[c] = load_k32(phin, D_, (c0 + c) * 16 + ln, k0, kh);
#pragma unroll
    for (int c = 0; c < 4; ++c) acc[c] = WMMA_BF16(a, b[c], acc[c]);
  }
  float* out = logits + (size_t)rowtile * 16 * NP_;
#pragma unroll
  for (int c = 0; c < 4; ++c)
#pragma unroll
    for (int r = 0; r < 8; ++r)
      out[(size_t)(r + 8 * kh) * NP_ + (c0 + c) * 16 + ln] = acc[c][r] * TEMP_INV;
}

// ---------------- softmaxes ----------------------------------------------

// combine: softmax over the 128 slots for each (b,m); emit bf16 (GEMM4 A).
__global__ void smoe_softmax_c(const float* __restrict__ logits, __bf16* __restrict__ cb) {
  int row = blockIdx.x, t = threadIdx.x;                 // 128 threads
  float v = logits[(size_t)row * NP_ + t];
  __shared__ float red[128];
  red[t] = v; __syncthreads();
  for (int s = 64; s > 0; s >>= 1) { if (t < s) red[t] = fmaxf(red[t], red[t + s]); __syncthreads(); }
  float mx = red[0]; __syncthreads();
  float e = __expf(v - mx);
  red[t] = e; __syncthreads();
  for (int s = 64; s > 0; s >>= 1) { if (t < s) red[t] += red[t + s]; __syncthreads(); }
  cb[(size_t)row * NP_ + t] = f2bf(e / red[0]);
}

// dispatch: softmax over the 4096 tokens for each (b,slot); emit transposed
// bf16 dt[b][slot][m] so GEMM2's A operand is K(m)-contiguous.
__global__ void smoe_softmax_d(const float* __restrict__ logits, __bf16* __restrict__ dt) {
  int b = blockIdx.x >> 7, s = blockIdx.x & 127, t = threadIdx.x;  // 256 thr
  const float* base = logits + (size_t)b * M_ * NP_ + s;
  float v[16], mx = -1e30f;
#pragma unroll
  for (int i = 0; i < 16; ++i) { v[i] = base[(size_t)(t + 256 * i) * NP_]; mx = fmaxf(mx, v[i]); }
  __shared__ float red[256];
  red[t] = mx; __syncthreads();
  for (int q = 128; q > 0; q >>= 1) { if (t < q) red[t] = fmaxf(red[t], red[t + q]); __syncthreads(); }
  mx = red[0]; __syncthreads();
  float sum = 0.f;
#pragma unroll
  for (int i = 0; i < 16; ++i) { v[i] = __expf(v[i] - mx); sum += v[i]; }
  red[t] = sum; __syncthreads();
  for (int q = 128; q > 0; q >>= 1) { if (t < q) red[t] += red[t + q]; __syncthreads(); }
  float inv = 1.0f / red[0];
  __bf16* o = dt + ((size_t)b * NP_ + s) * M_;
#pragma unroll
  for (int i = 0; i < 16; ++i) o[t + 256 * i] = f2bf(v[i] * inv);
}

// ---------------- GEMM2: xs = d^T @ x  (per batch, K=4096) ----------------
// One wave: 16 slots x 64 d-cols (4 tiles, A reused); per k-step:
// 10-load clause then 4 WMMAs.
__global__ void smoe_xs(const __bf16* __restrict__ dt, const __bf16* __restrict__ xT,
                        __bf16* __restrict__ xs) {
  int d64 = blockIdx.x, stile = blockIdx.y, b = blockIdx.z;
  int lane = threadIdx.x & 31, ln = lane & 15, kh = lane >> 4;
  const __bf16* A  = dt + ((size_t)b * NP_ + stile * 16) * M_;
  const __bf16* Bm = xT + ((size_t)b * D_  + d64 * 64) * M_;
  v8f zero = {};
  v8f acc[4];
#pragma unroll
  for (int c = 0; c < 4; ++c) acc[c] = zero;
  for (int k0 = 0; k0 < M_; k0 += 32) {
    v16bf a = load_k32(A, M_, ln, k0, kh);
    v16bf b[4];
#pragma unroll
    for (int c = 0; c < 4; ++c) b[c] = load_k32(Bm, M_, c * 16 + ln, k0, kh);
#pragma unroll
    for (int c = 0; c < 4; ++c) acc[c] = WMMA_BF16(a, b[c], acc[c]);
  }
  __bf16* o = xs + ((size_t)b * NP_ + stile * 16) * D_ + d64 * 64;
#pragma unroll
  for (int c = 0; c < 4; ++c)
#pragma unroll
    for (int r = 0; r < 8; ++r)
      o[(size_t)(r + 8 * kh) * D_ + c * 16 + ln] = f2bf(acc[c][r]);
}

// ---------------- GEMM3: ys = xs @ W[n] + b, stored transposed ------------
// Per expert n: 16 rows (b,p) x 16 cols; K=1024, k-unrolled x2:
// 8-load clause (two k-slices of A and B) then 2 WMMAs.
// Output yst[b][e][n*2+p] (GEMM4 B operand).
__global__ void smoe_ys(const __bf16* __restrict__ xs, const __bf16* __restrict__ Wt,
                        const float* __restrict__ bias, __bf16* __restrict__ yst) {
  int etile = blockIdx.x, n = blockIdx.y;
  int lane = threadIdx.x & 31, ln = lane & 15, kh = lane >> 4;
  int bb = ln >> 1, p = ln & 1;                 // A row m = bb*2+p
  int arow = bb * NP_ + n * 2 + p;              // row into xs[B*NP][D]
  const __bf16* Bm = Wt + (size_t)n * D_ * D_ + (size_t)etile * 16 * D_;
  v8f acc = {};
  for (int k0 = 0; k0 < D_; k0 += 64) {
    v16bf a0 = load_k32(xs, D_, arow, k0,      kh);
    v16bf b0 = load_k32(Bm, D_, ln,   k0,      kh);
    v16bf a1 = load_k32(xs, D_, arow, k0 + 32, kh);
    v16bf b1 = load_k32(Bm, D_, ln,   k0 + 32, kh);
    acc = WMMA_BF16(a0, b0, acc);
    acc = WMMA_BF16(a1, b1, acc);
  }
  int e = etile * 16 + ln;
  float bv = bias[(size_t)n * D_ + e];
#pragma unroll
  for (int r = 0; r < 8; ++r) {
    int m = r + 8 * kh, b2 = m >> 1, p2 = m & 1;
    yst[((size_t)b2 * D_ + e) * NP_ + n * 2 + p2] = f2bf(acc[r] + bv);
  }
}

// ---------------- GEMM4: y = c @ ys + x  (per batch, K=128) ---------------
// One wave: 16 m-rows x 64 d-cols; A (all of K) loaded once, K unrolled.
__global__ void smoe_y(const __bf16* __restrict__ cb, const __bf16* __restrict__ yst,
                       const float* __restrict__ x, float* __restrict__ y) {
  int d64 = blockIdx.x, mtile = blockIdx.y, b = blockIdx.z;
  int lane = threadIdx.x & 31, ln = lane & 15, kh = lane >> 4;
  const __bf16* A  = cb  + ((size_t)b * M_ + mtile * 16) * NP_;
  const __bf16* Bm = yst + ((size_t)b * D_ + d64 * 64) * NP_;
  v16bf a[4];
#pragma unroll
  for (int k = 0; k < 4; ++k) a[k] = load_k32(A, NP_, ln, k * 32, kh);
  const float* xr = x + ((size_t)b * M_ + mtile * 16) * D_ + d64 * 64;
  float*       yr = y + ((size_t)b * M_ + mtile * 16) * D_ + d64 * 64;
#pragma unroll
  for (int c = 0; c < 4; ++c) {
    v16bf bm[4];
#pragma unroll
    for (int k = 0; k < 4; ++k) bm[k] = load_k32(Bm, NP_, c * 16 + ln, k * 32, kh);
    v8f acc = {};
#pragma unroll
    for (int k = 0; k < 4; ++k) acc = WMMA_BF16(a[k], bm[k], acc);
#pragma unroll
    for (int r = 0; r < 8; ++r) {
      int m = r + 8 * kh;
      yr[(size_t)m * D_ + c * 16 + ln] = acc[r] + xr[(size_t)m * D_ + c * 16 + ln];
    }
  }
}

// ---------------- launch ---------------------------------------------------
extern "C" void kernel_launch(void* const* d_in, const int* in_sizes, int n_in,
                              void* d_out, int out_size, void* d_ws, size_t ws_size,
                              hipStream_t stream) {
  const float* x     = (const float*)d_in[0];
  const float* phi   = (const float*)d_in[1];
  const float* scale = (const float*)d_in[2];
  const float* W     = (const float*)d_in[3];
  const float* bias  = (const float*)d_in[4];
  float* y = (float*)d_out;

  char* p = (char*)d_ws;
  auto alloc = [&](size_t bytes) -> char* {
    char* r = p; p += (bytes + 255) & ~(size_t)255; return r;
  };
  __bf16* xn     = (__bf16*)alloc((size_t)B_ * M_ * D_ * 2);   // 64 MB
  __bf16* xT     = (__bf16*)alloc((size_t)B_ * D_ * M_ * 2);   // 64 MB
  __bf16* phin   = (__bf16*)alloc((size_t)NP_ * D_ * 2);       // 256 KB
  float*  logits = (float*) alloc((size_t)B_ * M_ * NP_ * 4);  // 16 MB
  __bf16* cb     = (__bf16*)alloc((size_t)B_ * M_ * NP_ * 2);  // 8 MB
  __bf16* dt     = (__bf16*)alloc((size_t)B_ * NP_ * M_ * 2);  // 8 MB
  __bf16* xs     = (__bf16*)alloc((size_t)B_ * NP_ * D_ * 2);  // 2 MB
  __bf16* Wt     = (__bf16*)alloc((size_t)N_ * D_ * D_ * 2);   // 128 MB
  __bf16* yst    = (__bf16*)alloc((size_t)B_ * D_ * NP_ * 2);  // 2 MB

  smoe_norm_x   <<<B_ * M_, 256, 0, stream>>>(x, xn);
  smoe_prep_phi <<<NP_, 256, 0, stream>>>(phi, scale, phin);
  smoe_tr_x     <<<dim3(M_ / 32, D_ / 32, B_), dim3(32, 8), 0, stream>>>(x, xT);
  smoe_tr_w     <<<dim3(D_ / 32, D_ / 32, N_), dim3(32, 8), 0, stream>>>(W, Wt);
  smoe_logits   <<<(B_ * M_ / 16) / 2, 128, 0, stream>>>(xn, phin, logits);
  smoe_softmax_c<<<B_ * M_, 128, 0, stream>>>(logits, cb);
  smoe_softmax_d<<<B_ * NP_, 256, 0, stream>>>(logits, dt);
  smoe_xs       <<<dim3(D_ / 64, NP_ / 16, B_), 32, 0, stream>>>(dt, xT, xs);
  smoe_ys       <<<dim3(D_ / 16, N_), 32, 0, stream>>>(xs, Wt, bias, yst);
  smoe_y        <<<dim3(D_ / 64, M_ / 16, B_), 32, 0, stream>>>(cb, yst, x, y);
}
